// MLPWithAttention_12232066859085
// MI455X (gfx1250) — compile-verified, matrix operand order checked by reading, not run on hardware
//
#include <hip/hip_runtime.h>
#include <hip/hip_bf16.h>
#include <math.h>

typedef __bf16 bf16;
typedef __attribute__((ext_vector_type(16))) __bf16 v16bf;
typedef __attribute__((ext_vector_type(8)))  __bf16 bf16x8;
typedef __attribute__((ext_vector_type(8)))  float  v8f;

constexpr int BM = 128, BN = 64, BK = 32, PAD = 8, LDSW = BK + PAD;

// ---------------------------------------------------------------------------
// Stage one (A:[128x32], B:[32x64]) tile pair into LDS as bf16.
// Phase 1 issues ALL global loads (clause-able, staged loadcnt waits);
// phase 2 converts + stores to LDS. B is stored transposed (Bs[n][k]) so
// B-fragment reads are contiguous in K.
// ---------------------------------------------------------------------------
template <typename InT>
__device__ __forceinline__
void stage_tiles(const InT* __restrict__ A, const InT* __restrict__ B,
                 int lda, int ldb, int m0, int n0, int k0, int tid,
                 bf16 (*As)[LDSW], bf16 (*Bs)[LDSW])
{
    if constexpr (sizeof(InT) == 4) {
        float4 va[4], vb[2];
        #pragma unroll
        for (int i = 0; i < 4; ++i) {                  // A: 128 rows x 8 float4
            int idx = tid + i * 256;
            int r = idx >> 3, c4 = (idx & 7) * 4;
            va[i] = *(const float4*)(A + (long long)(m0 + r) * lda + k0 + c4);
        }
        #pragma unroll
        for (int i = 0; i < 2; ++i) {                  // B: 32 rows x 16 float4
            int idx = tid + i * 256;
            int r = idx >> 4, c4 = (idx & 15) * 4;
            vb[i] = *(const float4*)(B + (long long)(k0 + r) * ldb + n0 + c4);
        }
        #pragma unroll
        for (int i = 0; i < 4; ++i) {
            int idx = tid + i * 256;
            int r = idx >> 3, c4 = (idx & 7) * 4;
            bf16* d = &As[r][c4];
            d[0] = (bf16)va[i].x; d[1] = (bf16)va[i].y;
            d[2] = (bf16)va[i].z; d[3] = (bf16)va[i].w;
        }
        #pragma unroll
        for (int i = 0; i < 2; ++i) {
            int idx = tid + i * 256;
            int r = idx >> 4, c4 = (idx & 15) * 4;
            Bs[c4 + 0][r] = (bf16)vb[i].x; Bs[c4 + 1][r] = (bf16)vb[i].y;
            Bs[c4 + 2][r] = (bf16)vb[i].z; Bs[c4 + 3][r] = (bf16)vb[i].w;
        }
    } else {
        bf16x8 va[2], vbv;
        #pragma unroll
        for (int i = 0; i < 2; ++i) {                  // A: 128 rows x 4 chunks of 8
            int idx = tid + i * 256;
            int r = idx >> 2, c8 = (idx & 3) * 8;
            va[i] = *(const bf16x8*)(A + (long long)(m0 + r) * lda + k0 + c8);
        }
        {                                              // B: 32 rows x 8 chunks of 8
            int r = tid >> 3, c8 = (tid & 7) * 8;
            vbv = *(const bf16x8*)(B + (long long)(k0 + r) * ldb + n0 + c8);
        }
        #pragma unroll
        for (int i = 0; i < 2; ++i) {
            int idx = tid + i * 256;
            int r = idx >> 2, c8 = (idx & 3) * 8;
            *(bf16x8*)&As[r][c8] = va[i];
        }
        {
            int r = tid >> 3, c8 = (tid & 7) * 8;
            #pragma unroll
            for (int e = 0; e < 8; ++e) Bs[c8 + e][r] = vbv[e];
        }
    }
}

// 16x32 bf16 A-fragment per ISA layout: lane m<16 -> K 0..7 & 16..23,
// lane m+16 -> K 8..15 & 24..31.  Same pattern serves B from its [n][k] store.
__device__ __forceinline__
v16bf load_frag(const bf16 (*T)[LDSW], int row, int half)
{
    bf16x8 lo = *(const bf16x8*)&T[row][half * 8];
    bf16x8 hi = *(const bf16x8*)&T[row][16 + half * 8];
    v16bf f;
    #pragma unroll
    for (int j = 0; j < 8; ++j) { f[j] = lo[j]; f[j + 8] = hi[j]; }
    return f;
}

// ---------------------------------------------------------------------------
// Double-buffered bf16-WMMA GEMM:  C[M,N] = alpha * A[M,K] * B[K,N]
// Block = 256 threads (8 waves), tile 128x64x32. Each wave owns a 32x32
// output patch: 2 A-frags x 2 B-frags -> 4 v_wmma_f32_16x16x32_bf16 / K-step,
// single barrier per K-step (stage next tile into alternate LDS buffer).
// TRANS_OUT stores C transposed (produces K^T directly).
// ---------------------------------------------------------------------------
template <typename InT, typename OutT, bool TRANS_OUT>
__global__ __launch_bounds__(256)
void gemm_wmma_kernel(const InT* __restrict__ Ag, const InT* __restrict__ Bg,
                      OutT* __restrict__ Cg,
                      int M, int N, int K, int lda, int ldb, int ldc,
                      long long strideA, long long strideB, long long strideC,
                      float alpha)
{
    __shared__ __attribute__((aligned(16))) bf16 As[2][BM][LDSW];
    __shared__ __attribute__((aligned(16))) bf16 Bs[2][BN][LDSW];

    const int tid  = threadIdx.x;
    const int wave = tid >> 5;
    const int lane = tid & 31;
    const int half = lane >> 4;
    const int l16  = lane & 15;

    const long long zb = blockIdx.z;
    const InT* A = Ag + zb * strideA;
    const InT* B = Bg + zb * strideB;
    OutT*      C = Cg + zb * strideC;

    const int m0 = blockIdx.y * BM;
    const int n0 = blockIdx.x * BN;

    const int wm = (wave & 3) * 32;    // wave's 32-row stripe in tile
    const int wn = (wave >> 2) * 32;   // wave's 32-col stripe in tile

    v8f acc00 = {}, acc01 = {}, acc10 = {}, acc11 = {};

    stage_tiles<InT>(A, B, lda, ldb, m0, n0, 0, tid, As[0], Bs[0]);

    int buf = 0;
    for (int k0 = 0; k0 < K; k0 += BK, buf ^= 1) {
        __syncthreads();                       // staged tile 'buf' visible
        if (k0 + BK < K) {
            stage_tiles<InT>(A, B, lda, ldb, m0, n0, k0 + BK, tid,
                             As[buf ^ 1], Bs[buf ^ 1]);
            if (k0 + 2 * BK < K) {             // gfx1250 global_prefetch_b8
                __builtin_prefetch(A + (long long)(m0 + (tid >> 2)) * lda + k0 + 2 * BK, 0, 1);
                __builtin_prefetch(B + (long long)(k0 + 2 * BK + (tid >> 3)) * ldb + n0, 0, 1);
            }
        }

        v16bf a0 = load_frag(As[buf], wm + l16,      half);
        v16bf a1 = load_frag(As[buf], wm + 16 + l16, half);
        v16bf b0 = load_frag(Bs[buf], wn + l16,      half);
        v16bf b1 = load_frag(Bs[buf], wn + 16 + l16, half);

        acc00 = __builtin_amdgcn_wmma_f32_16x16x32_bf16(false, a0, false, b0, (short)0, acc00, false, false);
        acc01 = __builtin_amdgcn_wmma_f32_16x16x32_bf16(false, a0, false, b1, (short)0, acc01, false, false);
        acc10 = __builtin_amdgcn_wmma_f32_16x16x32_bf16(false, a1, false, b0, (short)0, acc10, false, false);
        acc11 = __builtin_amdgcn_wmma_f32_16x16x32_bf16(false, a1, false, b1, (short)0, acc11, false, false);
    }

    // C layout: VGPR r -> M = r + 8*half, N = lane%16 (per 16x16 f32 C table)
    #pragma unroll
    for (int tm = 0; tm < 2; ++tm) {
        #pragma unroll
        for (int tn = 0; tn < 2; ++tn) {
            const v8f& acc = tm == 0 ? (tn == 0 ? acc00 : acc01)
                                     : (tn == 0 ? acc10 : acc11);
            #pragma unroll
            for (int r = 0; r < 8; ++r) {
                int m = m0 + wm + tm * 16 + r + half * 8;
                int n = n0 + wn + tn * 16 + l16;
                float v = acc[r] * alpha;
                if constexpr (TRANS_OUT)
                    C[(long long)n * ldc + m] = (OutT)v;
                else
                    C[(long long)m * ldc + n] = (OutT)v;
            }
        }
    }
}

// ---------------------------------------------------------------------------
// Row softmax over Lc=2048 bf16 values, in place. One block per (b,s) row.
// ---------------------------------------------------------------------------
__global__ __launch_bounds__(256)
void softmax_kernel(bf16* __restrict__ S, int Lc)
{
    const long long row = blockIdx.x;
    bf16* p = S + row * (long long)Lc;
    const int tid = threadIdx.x;
    constexpr int PER = 8;          // 2048 / 256

    float v[PER];
    float mx = -1e30f;
    #pragma unroll
    for (int i = 0; i < PER; ++i) {
        v[i] = (float)p[tid + i * 256];
        mx = fmaxf(mx, v[i]);
    }
    __shared__ float red[256];
    red[tid] = mx; __syncthreads();
    for (int s = 128; s > 0; s >>= 1) {
        if (tid < s) red[tid] = fmaxf(red[tid], red[tid + s]);
        __syncthreads();
    }
    mx = red[0]; __syncthreads();

    float sum = 0.f;
    #pragma unroll
    for (int i = 0; i < PER; ++i) { v[i] = __expf(v[i] - mx); sum += v[i]; }
    red[tid] = sum; __syncthreads();
    for (int s = 128; s > 0; s >>= 1) {
        if (tid < s) red[tid] += red[tid + s];
        __syncthreads();
    }
    const float inv = 1.0f / red[0];
    #pragma unroll
    for (int i = 0; i < PER; ++i) p[tid + i * 256] = (bf16)(v[i] * inv);
}

// ---------------------------------------------------------------------------
// x[b, 0:D]  = mean_s O[b,s,:]    x[b, D:2D] = mean_s substrate[b,s,:]
// ---------------------------------------------------------------------------
__global__ __launch_bounds__(128)
void mean_concat_kernel(const float* __restrict__ O, const float* __restrict__ sub,
                        float* __restrict__ X, int Ls, int D)
{
    const int b = blockIdx.y;
    const int j = blockIdx.x * 128 + threadIdx.x;     // 0 .. 2D-1
    const float* src;
    int col;
    if (j < D) { src = O   + (long long)b * Ls * D; col = j;     }
    else       { src = sub + (long long)b * Ls * D; col = j - D; }
    float s = 0.f;
    for (int t = 0; t < Ls; ++t) s += src[(long long)t * D + col];
    X[b * 2 * D + j] = s * (1.0f / (float)Ls);
}

// ---------------------------------------------------------------------------
// MLP head: relu(x@W1+b1) -> relu(@W2+b2) -> sigmoid(@W3+b3). One block per b.
// ---------------------------------------------------------------------------
__global__ __launch_bounds__(256)
void mlp_kernel(const float* __restrict__ X,
                const float* __restrict__ W1, const float* __restrict__ b1,
                const float* __restrict__ W2, const float* __restrict__ b2,
                const float* __restrict__ W3, const float* __restrict__ b3,
                float* __restrict__ out)
{
    const int b = blockIdx.x, tid = threadIdx.x;
    __shared__ float x[2560];
    __shared__ float h1[512];
    __shared__ float h2[64];

    for (int i = tid; i < 2560; i += 256) x[i] = X[b * 2560 + i];
    __syncthreads();

    for (int n = tid; n < 512; n += 256) {
        float s = b1[n];
        for (int i = 0; i < 2560; ++i) s += x[i] * W1[i * 512 + n];
        h1[n] = fmaxf(s, 0.0f);
    }
    __syncthreads();

    if (tid < 64) {
        float s = b2[tid];
        for (int i = 0; i < 512; ++i) s += h1[i] * W2[i * 64 + tid];
        h2[tid] = fmaxf(s, 0.0f);
    }
    __syncthreads();

    if (tid == 0) {
        float s = b3[0];
        for (int i = 0; i < 64; ++i) s += h2[i] * W3[i];
        out[b] = 1.0f / (1.0f + __expf(-s));
    }
}

// ---------------------------------------------------------------------------
extern "C" void kernel_launch(void* const* d_in, const int* in_sizes, int n_in,
                              void* d_out, int out_size, void* d_ws, size_t ws_size,
                              hipStream_t stream)
{
    (void)in_sizes; (void)n_in; (void)out_size; (void)ws_size;

    const float* cyclase   = (const float*)d_in[0];
    const float* substrate = (const float*)d_in[1];
    // d_in[2], d_in[3]: masks — reference bug makes them no-ops; ignored.
    const float* Wq = (const float*)d_in[4];
    const float* Wk = (const float*)d_in[5];
    const float* Wv = (const float*)d_in[6];
    const float* W1 = (const float*)d_in[7];
    const float* b1 = (const float*)d_in[8];
    const float* W2 = (const float*)d_in[9];
    const float* b2 = (const float*)d_in[10];
    const float* W3 = (const float*)d_in[11];
    const float* b3 = (const float*)d_in[12];
    float* out = (float*)d_out;

    const int B = 16, Ls = 1024, Lc = 2048, D = 1280;

    char*  ws  = (char*)d_ws;
    size_t off = 0;
    auto alloc = [&](size_t bytes) {
        void* p = ws + off;
        off += (bytes + 255) & ~(size_t)255;
        return p;
    };
    bf16*  Q  = (bf16*)alloc((size_t)B * Ls * D * 2);   //  42 MB
    bf16*  Kt = (bf16*)alloc((size_t)B * D * Lc * 2);   //  84 MB  (K^T per batch)
    bf16*  V  = (bf16*)alloc((size_t)B * Lc * D * 2);   //  84 MB
    bf16*  S  = (bf16*)alloc((size_t)B * Ls * Lc * 2);  //  67 MB  (scores -> probs in place)
    float* O  = (float*)alloc((size_t)B * Ls * D * 4);  //  84 MB
    float* X  = (float*)alloc((size_t)B * 2 * D * 4);   // tiny

    const dim3 blk(256);
    const float inv_sqrt_d = 1.0f / sqrtf((float)D);

    // 1) Q = substrate @ Wq          (batch flattened: M = B*Ls)
    gemm_wmma_kernel<float, bf16, false><<<dim3(D / BN, (B * Ls) / BM, 1), blk, 0, stream>>>(
        substrate, Wq, Q, B * Ls, D, D, D, D, D, 0, 0, 0, 1.0f);

    // 2) Kt[b] = (cyclase[b] @ Wk)^T  -> [D][Lc]
    gemm_wmma_kernel<float, bf16, true><<<dim3(D / BN, Lc / BM, B), blk, 0, stream>>>(
        cyclase, Wk, Kt, Lc, D, D, D, D, Lc,
        (long long)Lc * D, 0, (long long)D * Lc, 1.0f);

    // 3) V = cyclase @ Wv            (batch flattened)
    gemm_wmma_kernel<float, bf16, false><<<dim3(D / BN, (B * Lc) / BM, 1), blk, 0, stream>>>(
        cyclase, Wv, V, B * Lc, D, D, D, D, D, 0, 0, 0, 1.0f);

    // 4) S[b] = (Q[b] @ Kt[b]) / sqrt(D)
    gemm_wmma_kernel<bf16, bf16, false><<<dim3(Lc / BN, Ls / BM, B), blk, 0, stream>>>(
        Q, Kt, S, Ls, Lc, D, D, Lc, Lc,
        (long long)Ls * D, (long long)D * Lc, (long long)Ls * Lc, inv_sqrt_d);

    // 5) softmax over Lc, in place (S -> P)
    softmax_kernel<<<dim3(B * Ls), blk, 0, stream>>>(S, Lc);

    // 6) O[b] = P[b] @ V[b]
    gemm_wmma_kernel<bf16, float, false><<<dim3(D / BN, Ls / BM, B), blk, 0, stream>>>(
        S, V, O, Ls, D, Lc, Lc, D, D,
        (long long)Ls * Lc, (long long)Lc * D, (long long)Ls * D, 1.0f);

    // 7) mean-pool + concat -> X[B, 2D]
    mean_concat_kernel<<<dim3((2 * D) / 128, B), dim3(128), 0, stream>>>(O, substrate, X, Ls, D);

    // 8) MLP head -> out[B]
    mlp_kernel<<<dim3(B), blk, 0, stream>>>(X, W1, b1, W2, b2, W3, b3, out);
}
